// ShiftPredictorWithRetrieval_40200893890573
// MI455X (gfx1250) — matrix operand order repaired and checked by reading, not compiled, not run on hardware
//
#include <hip/hip_runtime.h>
#include <hip/hip_bf16.h>
#include <math.h>

// ---------------------------------------------------------------------------
// ShiftPredictorWithRetrieval for MI455X (gfx1250, wave32, WMMA)
// Heavy GEMMs: LDS-staged, double-buffered v_wmma_f32_16x16x32_bf16,
// async global->LDS staging when the toolchain exposes the builtin.
// ---------------------------------------------------------------------------

typedef __attribute__((ext_vector_type(16))) __bf16 v16bf;
typedef __attribute__((ext_vector_type(8)))  float  v8f;
typedef int async_v4i __attribute__((ext_vector_type(4)));

#define B_   8192
#define K_   32
#define S_   6
#define QD_  1024
#define H_   512
#define NH_  4
#define HD_  128
#define NR_  20

#if __has_builtin(__builtin_amdgcn_global_load_async_to_lds_b128) && \
    __has_builtin(__builtin_amdgcn_s_wait_asynccnt)
#define HAVE_ASYNC_LDS 1
#else
#define HAVE_ASYNC_LDS 0
#endif

// exact vector-pointer types the async builtin expects (from hipcc diagnostic)
#define AS1_V4I(p) ((async_v4i __attribute__((address_space(1)))*)(void*)(p))
#define AS3_V4I(p) ((async_v4i __attribute__((address_space(3)))*)(void*)(p))

__device__ __forceinline__ float gelu_exact(float x) {
  return 0.5f * x * (1.0f + erff(x * 0.70710678118654752440f));
}

// ---------------------------------------------------------------------------
// Weight transpose + f32->bf16 convert:  W [din, dout] -> Wt [dout, dinPad]
// ---------------------------------------------------------------------------
__global__ void k_transpose_wt(const float* __restrict__ W, __bf16* __restrict__ Wt,
                               int din, int dout, int dinPad) {
  long i = (long)blockIdx.x * blockDim.x + threadIdx.x;
  long total = (long)dout * dinPad;
  if (i >= total) return;
  int n = (int)(i / dinPad);
  int k = (int)(i % dinPad);
  float v = (k < din) ? W[(long)k * dout + n] : 0.0f;
  Wt[i] = (__bf16)v;
}

// flat f32 -> bf16 convert
__global__ void k_f32_to_bf16(const float* __restrict__ X, __bf16* __restrict__ Y, long n) {
  long i = (long)blockIdx.x * blockDim.x + threadIdx.x;
  if (i < n) Y[i] = (__bf16)X[i];
}

// ---------------------------------------------------------------------------
// Random-coil correction, same_type, and padded bf16 context-proj input
// ---------------------------------------------------------------------------
__global__ void k_prep(const int* __restrict__ qcode, const int* __restrict__ rcodes,
                       const float* __restrict__ rshifts, const unsigned char* __restrict__ rmasks,
                       const float* __restrict__ rdist, const float* __restrict__ rc_table,
                       const float* __restrict__ res_emb,
                       float* __restrict__ shifts, float* __restrict__ same_t,
                       __bf16* __restrict__ ctx_in) {
  int i = blockIdx.x * blockDim.x + threadIdx.x;
  if (i >= B_ * K_) return;
  int b = i >> 5;
  int qc = qcode[b], rc = rcodes[i];
  int qci = qc < 0 ? 0 : (qc > NR_ ? NR_ : qc);
  int rci = rc < 0 ? 0 : (rc > NR_ ? NR_ : rc);
  float st = (rc == qc) ? 1.0f : 0.0f;
  same_t[i] = st;
  __bf16* ci = ctx_in + (long)i * 32;
  ci[0] = (__bf16)rdist[i];
  ci[1] = (__bf16)st;
#pragma unroll
  for (int j = 0; j < 16; ++j) ci[2 + j] = (__bf16)res_emb[rci * 16 + j];
#pragma unroll
  for (int j = 18; j < 32; ++j) ci[j] = (__bf16)0.0f;
#pragma unroll
  for (int s = 0; s < S_; ++s) {
    float rq = rc_table[qci * S_ + s];
    float rr = rc_table[rci * S_ + s];
    float sh = rshifts[(long)i * S_ + s];
    bool msk = rmasks[(long)i * S_ + s] != 0;
    bool rcv = (rq == rq) && (rr == rr);   // !isnan
    shifts[(long)i * S_ + s] = (rcv && msk) ? (rq + sh - rr) : sh;
  }
}

// ---------------------------------------------------------------------------
// Per-row global stats + any_valid
// ---------------------------------------------------------------------------
__global__ void k_gstats(const unsigned char* __restrict__ rvalid, const float* __restrict__ rdist,
                         const float* __restrict__ same_t, float* __restrict__ anyv,
                         float* __restrict__ gstats) {
  int b = blockIdx.x * blockDim.x + threadIdx.x;
  if (b >= B_) return;
  float cnt = 0.f, sumd = 0.f, maxd = -1e30f, sumst = 0.f;
  for (int k = 0; k < K_; ++k) {
    int bk = b * K_ + k;
    float v = rvalid[bk] ? 1.f : 0.f;
    float d = rdist[bk];
    cnt   += v;
    sumd  += v * d;
    maxd   = fmaxf(maxd, v * d);
    sumst += v * same_t[bk];
  }
  float c1 = fmaxf(cnt, 1.0f);
  anyv[b] = (cnt > 0.f) ? 1.f : 0.f;
  gstats[b * 4 + 0] = cnt / (float)K_;
  gstats[b * 4 + 1] = sumd / c1;
  gstats[b * 4 + 2] = maxd;
  gstats[b * 4 + 3] = sumst / c1;
}

// ---------------------------------------------------------------------------
// LDS-staged WMMA GEMM: C[M,N] = act(A[M,Kp] @ Bt[N,Kp]^T + bias)
// Block tile 64(M) x 128(N), K-step 32, double-buffered LDS staging.
// 8 waves = 2(M) x 4(N); each wave computes a 32x32 register tile (2x2 WMMA).
// ---------------------------------------------------------------------------
#define BM 64
#define BN 128
#define BK 32

__global__ __launch_bounds__(256) void gemm_bf16_wmma(
    const __bf16* __restrict__ A, const __bf16* __restrict__ Bt,
    const float* __restrict__ bias, float* __restrict__ C,
    int M, int N, int Kp, int act) {
  __shared__ __bf16 sA[2][BM * BK];   // 2 x 4KB
  __shared__ __bf16 sB[2][BN * BK];   // 2 x 8KB

  const int tid  = threadIdx.x;
  const int lane = tid & 31;
  const int wave = tid >> 5;
  const int wm   = wave & 1;          // 0..1 -> 32-row halves
  const int wn   = wave >> 1;         // 0..3 -> 32-col quarters
  const int half = lane >> 4;
  const int l15  = lane & 15;

  const long blockM = (long)blockIdx.x * BM;
  const long blockN = (long)blockIdx.y * BN;

  // Cooperative staging split: A tile 64x64B -> 16B/thread, B tile 128x64B -> 32B/thread
  const int arow = tid >> 2, aoff = (tid & 3) * 8;    // element offsets (8 bf16 = 16B)
  const int brow = tid >> 1, boff = (tid & 1) * 16;   // (16 bf16 = 32B)

  const __bf16* gA = A  + (blockM + arow) * (long)Kp + aoff;
  const __bf16* gB = Bt + (blockN + brow) * (long)Kp + boff;

  auto stage = [&](int bufi, int kk) {
    const __bf16* ga = gA + kk;
    const __bf16* gb = gB + kk;
    __bf16* la = &sA[bufi][arow * BK + aoff];
    __bf16* lb = &sB[bufi][brow * BK + boff];
#if HAVE_ASYNC_LDS
    __builtin_amdgcn_global_load_async_to_lds_b128(AS1_V4I(ga),     AS3_V4I(la),     0, 0);
    __builtin_amdgcn_global_load_async_to_lds_b128(AS1_V4I(gb),     AS3_V4I(lb),     0, 0);
    __builtin_amdgcn_global_load_async_to_lds_b128(AS1_V4I(gb + 8), AS3_V4I(lb + 8), 0, 0);
#else
    uint4 a0 = *(const uint4*)ga;
    uint4 b0 = *(const uint4*)gb;
    uint4 b1 = *(const uint4*)(gb + 8);
    *(uint4*)la       = a0;
    *(uint4*)lb       = b0;
    *(uint4*)(lb + 8) = b1;
#endif
  };

  v8f c00 = {}, c01 = {}, c10 = {}, c11 = {};
  union Frag { v16bf v; uint4 q[2]; };

  stage(0, 0);                       // preload first K-slice
  int buf = 0;
  for (int k0 = 0; k0 < Kp; k0 += BK) {
#if HAVE_ASYNC_LDS
    __builtin_amdgcn_s_wait_asynccnt(0);   // my async LDS writes done
#endif
    __syncthreads();                       // everyone's staging visible
    if (k0 + BK < Kp) stage(buf ^ 1, k0 + BK);

    const __bf16* a0 = &sA[buf][(wm * 32 +  0 + l15) * BK];
    const __bf16* a1 = &sA[buf][(wm * 32 + 16 + l15) * BK];
    const __bf16* b0 = &sB[buf][(wn * 32 +  0 + l15) * BK];
    const __bf16* b1 = &sB[buf][(wn * 32 + 16 + l15) * BK];

    Frag fa0, fa1, fb0, fb1;
    // A fragment: 16B chunks at {half*16B, 32B + half*16B} within the 64B row
    fa0.q[0] = *(const uint4*)(a0 + half * 8);
    fa0.q[1] = *(const uint4*)(a0 + 16 + half * 8);
    fa1.q[0] = *(const uint4*)(a1 + half * 8);
    fa1.q[1] = *(const uint4*)(a1 + 16 + half * 8);
    // B fragment: 16B chunks at {half*16B, 32B + half*16B} within the 64B row
    fb0.q[0] = *(const uint4*)(b0 + half * 8);
    fb0.q[1] = *(const uint4*)(b0 + 16 + half * 8);
    fb1.q[0] = *(const uint4*)(b1 + half * 8);
    fb1.q[1] = *(const uint4*)(b1 + 16 + half * 8);

    c00 = __builtin_amdgcn_wmma_f32_16x16x32_bf16(false, fa0.v, false, fb0.v, (short)0, c00, false, false);
    c01 = __builtin_amdgcn_wmma_f32_16x16x32_bf16(false, fa0.v, false, fb1.v, (short)0, c01, false, false);
    c10 = __builtin_amdgcn_wmma_f32_16x16x32_bf16(false, fa1.v, false, fb0.v, (short)0, c10, false, false);
    c11 = __builtin_amdgcn_wmma_f32_16x16x32_bf16(false, fa1.v, false, fb1.v, (short)0, c11, false, false);
    buf ^= 1;
  }

  const long col0 = blockN + wn * 32 + l15;
  const long col1 = col0 + 16;
  const float bv0 = bias ? bias[col0] : 0.0f;
  const float bv1 = bias ? bias[col1] : 0.0f;
#pragma unroll
  for (int r = 0; r < 8; ++r) {
    long m0 = blockM + wm * 32 + half * 8 + r;   // tile row 0
    long m1 = m0 + 16;                           // tile row 1
    float v00 = c00[r] + bv0, v01 = c01[r] + bv1;
    float v10 = c10[r] + bv0, v11 = c11[r] + bv1;
    if (act) { v00 = gelu_exact(v00); v01 = gelu_exact(v01);
               v10 = gelu_exact(v10); v11 = gelu_exact(v11); }
    C[m0 * (long)N + col0] = v00;
    C[m0 * (long)N + col1] = v01;
    C[m1 * (long)N + col0] = v10;
    C[m1 * (long)N + col1] = v11;
  }
}

// ---------------------------------------------------------------------------
// Attention: one wave32 per (b, head); lane == neighbor index k (K_==32).
// ---------------------------------------------------------------------------
__global__ __launch_bounds__(256) void k_attention(
    const float* __restrict__ Qh, const float* __restrict__ Kh,
    const float* __restrict__ Vh, const unsigned char* __restrict__ valid,
    float* __restrict__ out) {
  __shared__ float s_aw[256];
  const int lane = threadIdx.x & 31;
  const int wave = threadIdx.x >> 5;
  const int task = blockIdx.x * 8 + wave;   // b * NH + h
  const int b = task >> 2;
  const int h = task & 3;

  const float* qp = Qh + (long)b * H_ + h * HD_;
  const float* kp = Kh + ((long)b * K_ + lane) * H_ + h * HD_;
  float s = 0.f;
  for (int d = 0; d < HD_; ++d) s += qp[d] * kp[d];
  s *= 0.08838834764831845f;                 // HD^-0.5
  if (!valid[b * K_ + lane]) s = -10000.0f;

  float m = s;
  for (int off = 16; off > 0; off >>= 1) m = fmaxf(m, __shfl_xor(m, off, 32));
  float e = __expf(s - m);
  float sum = e;
  for (int off = 16; off > 0; off >>= 1) sum += __shfl_xor(sum, off, 32);
  float aw = e / sum;

  s_aw[wave * 32 + lane] = aw;
  __syncthreads();

  float acc0 = 0.f, acc1 = 0.f, acc2 = 0.f, acc3 = 0.f;
  const float* vbase = Vh + (long)b * K_ * H_ + h * HD_ + lane;
  for (int k = 0; k < K_; ++k) {
    float a = s_aw[wave * 32 + k];
    const float* vp = vbase + (long)k * H_;
    acc0 += a * vp[0];
    acc1 += a * vp[32];
    acc2 += a * vp[64];
    acc3 += a * vp[96];
  }
  float* op = out + (long)b * H_ + h * HD_ + lane;
  op[0]  = acc0;
  op[32] = acc1;
  op[64] = acc2;
  op[96] = acc3;
}

// ---------------------------------------------------------------------------
// Narrow dense projection (tsw2)
// ---------------------------------------------------------------------------
__global__ void k_rowdot(const float* __restrict__ X, const float* __restrict__ W,
                         const float* __restrict__ bias, float* __restrict__ out,
                         int rows, int Kd, int Ncols) {
  long i = (long)blockIdx.x * blockDim.x + threadIdx.x;
  if (i >= (long)rows * Ncols) return;
  int r = (int)(i / Ncols), c = (int)(i % Ncols);
  float acc = bias[c];
  const float* x = X + (long)r * Kd;
  for (int k = 0; k < Kd; ++k) acc += x[k] * W[(long)k * Ncols + c];
  out[i] = acc;
}

// ---------------------------------------------------------------------------
// Trust-gated weighted combine over K + per-shift stats + fallback
// ---------------------------------------------------------------------------
__global__ void k_combine(const float* __restrict__ rdist, const float* __restrict__ same_t,
                          const unsigned char* __restrict__ rvalid, const unsigned char* __restrict__ rmasks,
                          const float* __restrict__ shifts, const float* __restrict__ swb,
                          const float* __restrict__ shift_scale, const float* __restrict__ fallback,
                          const float* __restrict__ anyv,
                          float* __restrict__ transferred, float* __restrict__ pss) {
  int i = blockIdx.x * blockDim.x + threadIdx.x;
  if (i >= B_ * S_) return;
  int b = i / S_, s = i % S_;
  float sw = swb[i] * 0.1f;

  float wv[K_], shv[K_];
  unsigned vmbits = 0;
  float maxw = -1e30f;
#pragma unroll
  for (int k = 0; k < K_; ++k) {
    int bk = b * K_ + k;
    bool vm = rvalid[bk] && (rmasks[(long)bk * S_ + s] != 0);
    float w = vm ? (rdist[bk] + same_t[bk] * 0.5f + sw) : -10000.0f;
    wv[k]  = w;
    shv[k] = shifts[(long)bk * S_ + s];
    if (vm) vmbits |= (1u << k);
    maxw = fmaxf(maxw, w);
  }
  float sum = 0.f;
#pragma unroll
  for (int k = 0; k < K_; ++k) { float e = __expf(wv[k] - maxw); wv[k] = e; sum += e; }
  float inv = 1.0f / sum;

  float tr = 0.f, cnt = 0.f, ssum = 0.f, dsum = 0.f, stsum = 0.f;
#pragma unroll
  for (int k = 0; k < K_; ++k) {
    tr += wv[k] * inv * shv[k];
    if (vmbits & (1u << k)) {
      cnt += 1.f; ssum += shv[k];
      dsum += rdist[b * K_ + k]; stsum += same_t[b * K_ + k];
    }
  }
  float c1 = fmaxf(cnt, 1.0f);
  float mean_s = ssum / c1;
  float var = 0.f;
#pragma unroll
  for (int k = 0; k < K_; ++k)
    if (vmbits & (1u << k)) { float d = shv[k] - mean_s; var += d * d; }
  var /= c1;

  pss[(long)i * 4 + 0] = cnt / (float)K_;
  pss[(long)i * 4 + 1] = var;
  pss[(long)i * 4 + 2] = dsum / c1;
  pss[(long)i * 4 + 3] = stsum / c1;
  transferred[i] = (anyv[b] > 0.f) ? tr * shift_scale[s] : fallback[s];
}

// ---------------------------------------------------------------------------
// Build trust-ctx input: [tq(512) | trust_res_embed(32) | global_stats(4) | pad] -> 576
// ---------------------------------------------------------------------------
__global__ void k_build_tctx(const float* __restrict__ tq, const float* __restrict__ tres_emb,
                             const int* __restrict__ qcode, const float* __restrict__ gstats,
                             __bf16* __restrict__ out) {
  long i = (long)blockIdx.x * blockDim.x + threadIdx.x;
  if (i >= (long)B_ * 576) return;
  int b = (int)(i / 576), j = (int)(i % 576);
  float v;
  if (j < 512) v = tq[(long)b * H_ + j];
  else if (j < 544) {
    int q = qcode[b]; q = q < 0 ? 0 : (q > NR_ ? NR_ : q);
    v = tres_emb[q * 32 + (j - 512)];
  } else if (j < 548) v = gstats[b * 4 + (j - 544)];
  else v = 0.f;
  out[i] = (__bf16)v;
}

// ---------------------------------------------------------------------------
// Build tps1 input: [tctx(512) | per_shift_stats(4) | pad] -> 544, rows = B*S
// ---------------------------------------------------------------------------
__global__ void k_build_tin(const float* __restrict__ tctx, const float* __restrict__ pss,
                            __bf16* __restrict__ out) {
  long i = (long)blockIdx.x * blockDim.x + threadIdx.x;
  if (i >= (long)B_ * S_ * 544) return;
  long r = i / 544; int j = (int)(i % 544);
  int b = (int)(r / S_);
  float v;
  if (j < 512) v = tctx[(long)b * H_ + j];
  else if (j < 516) v = pss[r * 4 + (j - 512)];
  else v = 0.f;
  out[i] = (__bf16)v;
}

// ---------------------------------------------------------------------------
// tps2 + sigmoid + any_valid gate
// ---------------------------------------------------------------------------
__global__ void k_tps2(const float* __restrict__ h2, const float* __restrict__ w,
                       const float* __restrict__ bias, const float* __restrict__ anyv,
                       float* __restrict__ trust) {
  int i = blockIdx.x * blockDim.x + threadIdx.x;
  if (i >= B_ * S_) return;
  const float* x = h2 + (long)i * 256;
  float acc = bias[0];
  for (int k = 0; k < 256; ++k) acc += x[k] * w[k];
  float t = 1.0f / (1.0f + __expf(-acc));
  trust[i] = t * anyv[i / S_];
}

// ---------------------------------------------------------------------------
// Host orchestration
// ---------------------------------------------------------------------------
static inline unsigned gdim(long n, int b) { return (unsigned)((n + b - 1) / b); }

extern "C" void kernel_launch(void* const* d_in, const int* in_sizes, int n_in,
                              void* d_out, int out_size, void* d_ws, size_t ws_size,
                              hipStream_t stream) {
  (void)in_sizes; (void)n_in; (void)out_size; (void)ws_size;

  const int*   qcode   = (const int*)d_in[0];
  const float* rshifts = (const float*)d_in[1];
  const unsigned char* rmasks = (const unsigned char*)d_in[2];
  const int*   rcodes  = (const int*)d_in[3];
  const float* rdist   = (const float*)d_in[4];
  const unsigned char* rvalid = (const unsigned char*)d_in[5];
  const float* qenc    = (const float*)d_in[6];
  const float* rc_tab  = (const float*)d_in[7];
  const float* qp_w  = (const float*)d_in[8];   const float* qp_b  = (const float*)d_in[9];
  const float* res_e = (const float*)d_in[10];
  const float* cp_w  = (const float*)d_in[11];  const float* cp_b  = (const float*)d_in[12];
  const float* aq_w  = (const float*)d_in[13];  const float* aq_b  = (const float*)d_in[14];
  const float* ak_w  = (const float*)d_in[15];  const float* ak_b  = (const float*)d_in[16];
  const float* av_w  = (const float*)d_in[17];  const float* av_b  = (const float*)d_in[18];
  const float* t1_w  = (const float*)d_in[19];  const float* t1_b  = (const float*)d_in[20];
  const float* t2_w  = (const float*)d_in[21];  const float* t2_b  = (const float*)d_in[22];
  const float* tre   = (const float*)d_in[23];
  const float* tqp_w = (const float*)d_in[24];  const float* tqp_b = (const float*)d_in[25];
  const float* tcp_w = (const float*)d_in[26];  const float* tcp_b = (const float*)d_in[27];
  const float* p1_w  = (const float*)d_in[28];  const float* p1_b  = (const float*)d_in[29];
  const float* p2_w  = (const float*)d_in[30];  const float* p2_b  = (const float*)d_in[31];
  const float* sscale = (const float*)d_in[32];
  const float* fshift = (const float*)d_in[33];

  // ---- workspace carve (bump allocator, 256B aligned) ----
  char* p = (char*)d_ws;
  auto alloc = [&](size_t bytes) -> char* {
    char* r = p; p += (bytes + 255) & ~(size_t)255; return r;
  };
  __bf16* wt_qp   = (__bf16*)alloc((size_t)H_ * QD_ * 2);
  __bf16* wt_tqp  = (__bf16*)alloc((size_t)H_ * QD_ * 2);
  __bf16* wt_cp   = (__bf16*)alloc((size_t)H_ * 32  * 2);
  __bf16* wt_aq   = (__bf16*)alloc((size_t)H_ * H_  * 2);
  __bf16* wt_ak   = (__bf16*)alloc((size_t)H_ * H_  * 2);
  __bf16* wt_av   = (__bf16*)alloc((size_t)H_ * H_  * 2);
  __bf16* wt_t1   = (__bf16*)alloc((size_t)H_ * H_  * 2);
  __bf16* wt_tcp  = (__bf16*)alloc((size_t)H_ * 576 * 2);
  __bf16* wt_p1   = (__bf16*)alloc((size_t)256 * 544 * 2);
  __bf16* qe_bf   = (__bf16*)alloc((size_t)B_ * QD_ * 2);
  __bf16* ctx_in  = (__bf16*)alloc((size_t)B_ * K_ * 32 * 2);
  float*  shifts  = (float*)alloc((size_t)B_ * K_ * S_ * 4);
  float*  same_t  = (float*)alloc((size_t)B_ * K_ * 4);
  float*  anyv    = (float*)alloc((size_t)B_ * 4);
  float*  gstats  = (float*)alloc((size_t)B_ * 4 * 4);
  float*  q_f32   = (float*)alloc((size_t)B_ * H_ * 4);
  float*  tq_f32  = (float*)alloc((size_t)B_ * H_ * 4);
  float*  ctx_f32 = (float*)alloc((size_t)B_ * K_ * H_ * 4);   // later reused as Kh
  __bf16* q_bf    = (__bf16*)alloc((size_t)B_ * H_ * 2);
  __bf16* ctx_bf  = (__bf16*)alloc((size_t)B_ * K_ * H_ * 2);
  float*  Qh      = (float*)alloc((size_t)B_ * H_ * 4);
  float*  Vh      = (float*)alloc((size_t)B_ * K_ * H_ * 4);
  float*  Kh      = ctx_f32;                                    // alias (ctx dead by then)
  float*  aout    = (float*)alloc((size_t)B_ * H_ * 4);
  __bf16* aout_bf = (__bf16*)alloc((size_t)B_ * H_ * 2);
  float*  h1      = (float*)alloc((size_t)B_ * H_ * 4);
  float*  swb     = (float*)alloc((size_t)B_ * S_ * 4);
  float*  pss     = (float*)alloc((size_t)B_ * S_ * 4 * 4);
  __bf16* tctx_in = (__bf16*)alloc((size_t)B_ * 576 * 2);
  float*  tctx    = (float*)alloc((size_t)B_ * H_ * 4);
  __bf16* tin     = (__bf16*)alloc((size_t)B_ * S_ * 544 * 2);
  float*  h2      = (float*)alloc((size_t)B_ * S_ * 256 * 4);

  float* out_transfer = (float*)d_out;              // (B, S)
  float* out_trust    = out_transfer + B_ * S_;     // (B, S)

  const int T = 256;
  auto tw = [&](const float* W, __bf16* Wt, int din, int dout, int dinPad) {
    k_transpose_wt<<<gdim((long)dout * dinPad, T), T, 0, stream>>>(W, Wt, din, dout, dinPad);
  };
  auto gemm = [&](const __bf16* A, const __bf16* Bt, const float* bias, float* C,
                  int M, int N, int Kp, int act) {
    dim3 g(M / BM, N / BN);
    gemm_bf16_wmma<<<g, T, 0, stream>>>(A, Bt, bias, C, M, N, Kp, act);
  };

  // 1) weight prep
  tw(qp_w,  wt_qp,  QD_, H_,  QD_);
  tw(tqp_w, wt_tqp, QD_, H_,  QD_);
  tw(cp_w,  wt_cp,  18,  H_,  32);
  tw(aq_w,  wt_aq,  H_,  H_,  H_);
  tw(ak_w,  wt_ak,  H_,  H_,  H_);
  tw(av_w,  wt_av,  H_,  H_,  H_);
  tw(t1_w,  wt_t1,  H_,  H_,  H_);
  tw(tcp_w, wt_tcp, 548, H_,  576);
  tw(p1_w,  wt_p1,  516, 256, 544);

  // 2) activations / elementwise prep
  k_f32_to_bf16<<<gdim((long)B_ * QD_, T), T, 0, stream>>>(qenc, qe_bf, (long)B_ * QD_);
  k_prep<<<gdim((long)B_ * K_, T), T, 0, stream>>>(qcode, rcodes, rshifts, rmasks, rdist,
                                                   rc_tab, res_e, shifts, same_t, ctx_in);
  k_gstats<<<gdim(B_, T), T, 0, stream>>>(rvalid, rdist, same_t, anyv, gstats);

  // 3) projections (WMMA)
  gemm(qe_bf,  wt_qp,  qp_b,  q_f32,   B_,        H_, QD_, 1);   // q  = gelu(query_proj)
  gemm(qe_bf,  wt_tqp, tqp_b, tq_f32,  B_,        H_, QD_, 1);   // tq = gelu(trust_query_proj)
  gemm(ctx_in, wt_cp,  cp_b,  ctx_f32, B_ * K_,   H_, 32,  1);   // ctx = gelu(context_proj)

  k_f32_to_bf16<<<gdim((long)B_ * H_, T), T, 0, stream>>>(q_f32, q_bf, (long)B_ * H_);
  k_f32_to_bf16<<<gdim((long)B_ * K_ * H_, T), T, 0, stream>>>(ctx_f32, ctx_bf, (long)B_ * K_ * H_);

  gemm(q_bf,   wt_aq, aq_b, Qh, B_,      H_, H_, 0);             // Qh
  gemm(ctx_bf, wt_ak, ak_b, Kh, B_ * K_, H_, H_, 0);             // Kh (overwrites ctx_f32)
  gemm(ctx_bf, wt_av, av_b, Vh, B_ * K_, H_, H_, 0);             // Vh

  // 4) attention
  k_attention<<<B_ * NH_ / 8, T, 0, stream>>>(Qh, Kh, Vh, rvalid, aout);
  k_f32_to_bf16<<<gdim((long)B_ * H_, T), T, 0, stream>>>(aout, aout_bf, (long)B_ * H_);

  gemm(aout_bf, wt_t1, t1_b, h1, B_, H_, H_, 1);                 // gelu(tsw1)
  k_rowdot<<<gdim((long)B_ * S_, T), T, 0, stream>>>(h1, t2_w, t2_b, swb, B_, H_, S_);

  // 5) trust-gated combine + per-shift stats + transferred output
  k_combine<<<gdim((long)B_ * S_, T), T, 0, stream>>>(rdist, same_t, rvalid, rmasks, shifts,
                                                      swb, sscale, fshift, anyv,
                                                      out_transfer, pss);

  // 6) trust gate
  k_build_tctx<<<gdim((long)B_ * 576, T), T, 0, stream>>>(tq_f32, tre, qcode, gstats, tctx_in);
  gemm(tctx_in, wt_tcp, tcp_b, tctx, B_, H_, 576, 1);            // gelu(trust_ctx_proj)
  k_build_tin<<<gdim((long)B_ * S_ * 544, T), T, 0, stream>>>(tctx, pss, tin);
  gemm(tin, wt_p1, p1_b, h2, B_ * S_, 256, 544, 1);              // gelu(tps1)
  k_tps2<<<gdim((long)B_ * S_, T), T, 0, stream>>>(h2, p2_w, p2_b, anyv, out_trust);
}
